// MTGFLOWZL_81552839016484
// MI455X (gfx1250) — matrix operand order repaired and checked by reading, not compiled, not run on hardware
//
#include <hip/hip_runtime.h>

// ---------------------------------------------------------------------------
// Problem constants (B=128, K=51, L=60, D=1, H=32)
// ---------------------------------------------------------------------------
#define BB   128
#define KK_  51
#define LL   60
#define HH   32
#define KL   3060      // K*L
#define KL32 97920     // K*L*32
#define N1   500
#define NP   512       // padded 500 -> 512

#define LOG2PI_F 1.8378770664093453f

// d_out layout (floats): hid[128*100] | logp_mean[1] | h_gcn[128*97920] | logp[128]
#define HID_OFF  0
#define LPM_OFF  12800
#define HGCN_OFF 12801
#define LP_OFF   12546561

// workspace layout (floats)
#define WS_GRAPH 0                      // 128*51*51 = 332928
#define WS_H     332928                 // 12533760
#define WS_HGCNP 12866688               // 12533760
#define WS_Z1    25400448               // 128*512
#define WS_Z2    25465984               // 128*512
#define WS_ANN   25531520               // 128*128

// ---------------------------------------------------------------------------
// CDNA5 WMMA types / helpers
// ---------------------------------------------------------------------------
typedef __attribute__((ext_vector_type(16))) __bf16 v16bf;
typedef __attribute__((ext_vector_type(8)))  float  v8f;
typedef __attribute__((ext_vector_type(4)))  unsigned int v4u;
typedef __attribute__((ext_vector_type(8)))  int v8i;
typedef __attribute__((ext_vector_type(4)))  int v4i;

union AFrag { v16bf v; uint4 q[2]; unsigned short s[16]; };
union CFrag { v8f   v; float f[8]; };

static __device__ inline unsigned short f2bf(float f) {
    unsigned u = __float_as_uint(f);
    unsigned r = (u + 0x7FFFu + ((u >> 16) & 1u)) >> 16;  // round-to-nearest-even
    return (unsigned short)r;
}

static __device__ inline void czero(CFrag& c) {
#pragma unroll
    for (int i = 0; i < 8; i++) c.f[i] = 0.f;
}

// A fragment (16x32 bf16): lane m = lane&15; lanes<16 hold K 0..15,
// lanes>=16 hold K 16..31. `base` is row-major [rows][32] bf16.
static __device__ inline void load_afrag(AFrag& fr, const unsigned short* base,
                                         int row0, int lane) {
    int m  = lane & 15;
    int ks = (lane >> 4) * 16;
    const unsigned short* p = base + (size_t)(row0 + m) * 32 + ks;
    fr.q[0] = *(const uint4*)(p);
    fr.q[1] = *(const uint4*)(p + 8);
}

// B fragment (32x16 bf16): lane n = lane&15 selects column; lanes<16 hold
// K 0..15, lanes>=16 hold K 16..31. `baseT` is column-major [n][32] bf16.
static __device__ inline void load_bfrag(AFrag& fr, const unsigned short* baseT,
                                         int n0, int lane) {
    int n  = n0 + (lane & 15);
    int kb = (lane >> 4) * 16;
    const unsigned short* p = baseT + (size_t)n * 32 + kb;
    fr.q[0] = *(const uint4*)(p);
    fr.q[1] = *(const uint4*)(p + 8);
}

static __device__ inline float sigf(float x) { return 1.f / (1.f + __expf(-x)); }

// ---------------------------------------------------------------------------
// Tensor Data Mover: DMA a 2D fp32 tile (nrows x 32 cols, row stride ldb)
// from global memory into LDS at byte offset lds_off. tensor_dim0/1 are
// clamped to the valid remainder so OOB reads land as zeros in LDS (D# §8.2).
// D# group0/group1 per CDNA5 ISA.
// This toolchain exposes the 6-arg builtin:
//   (uint32x4 g0, int32x8 g1, int32x4 g2, int32x4 g3, int32x8 extra, i32 cpol)
// ---------------------------------------------------------------------------
static __device__ inline void tdm_load_tile_f32(const float* gptr, unsigned lds_off,
                                                int ncols_valid, int nrows_valid,
                                                int ldb_elems)
{
    unsigned long long ga = (unsigned long long)(uintptr_t)gptr;
    v4u g0;
    g0.x = 1u;                                   // count=1, is_restore=0
    g0.y = lds_off;                              // lds_addr (bytes)
    g0.z = (unsigned)(ga & 0xFFFFFFFFu);         // global_addr[31:0]
    g0.w = (unsigned)((ga >> 32) & 0x01FFFFFFu)  // global_addr[56:32]
         | (2u << 30);                           // type=2 ("image")
    v8i g1;
    g1[0] = (int)(2u << 16);                           // data_size=2 -> 4B elems
    g1[1] = (int)((unsigned)(ncols_valid & 0xFFFF) << 16); // tensor_dim0[15:0]
    g1[2] = (int)((unsigned)(nrows_valid & 0xFFFF) << 16); // tensor_dim1[15:0]
    g1[3] = (int)(32u << 16);                          // tile_dim0 = 32
    g1[4] = 32;                                        // tile_dim1 = 32
    g1[5] = ldb_elems;                                 // tensor_dim0_stride lo
    g1[6] = 0;
    g1[7] = 0;
    v4i gz = {0, 0, 0, 0};
    v8i gz8 = {0, 0, 0, 0, 0, 0, 0, 0};
    __builtin_amdgcn_tensor_load_to_lds(g0, g1, gz, gz, gz8, 0);
}

// ---------------------------------------------------------------------------
// K1: attention -> graph[b,i,j] = softmax_i( (q @ k.reshape(c,K))/sqrt(c) )
// one block per batch b
// ---------------------------------------------------------------------------
__global__ __launch_bounds__(256) void attn_kernel(
    const float* __restrict__ x,  const float* __restrict__ Wq,
    const float* __restrict__ bq, const float* __restrict__ Wk,
    const float* __restrict__ bk, float* __restrict__ graph)
{
    __shared__ float xf[KL], qL[KL], kL[KL], sc[KK_ * KK_];
    int tid = threadIdx.x, b = blockIdx.x;

    for (int e = tid; e < KL; e += 256) xf[e] = x[(size_t)b * KL + e];
    __syncthreads();

    for (int e = tid; e < KL; e += 256) {
        int i = e / 60, d2 = e % 60;
        float sq = bq[d2], sk = bk[d2];
        for (int d = 0; d < 60; d++) {
            float xv = xf[i * 60 + d];
            sq += xv * Wq[d * 60 + d2];
            sk += xv * Wk[d * 60 + d2];
        }
        qL[e] = sq; kL[e] = sk;
    }
    __syncthreads();

    const float scale = 0.12909944487358056f;  // 1/sqrt(60)
    for (int e = tid; e < KK_ * KK_; e += 256) {
        int i = e / 51, j = e % 51;
        float s = 0.f;
        // k_t = k.reshape(c, K): k_t[a][j] = kflat[a*51 + j]
        for (int a = 0; a < 60; a++) s += qL[i * 60 + a] * kL[a * 51 + j];
        sc[e] = s * scale;
    }
    __syncthreads();

    if (tid < 51) {  // softmax over axis i (axis=1 of (B,K,K))
        int j = tid;
        float mx = -1e30f;
        for (int i = 0; i < 51; i++) mx = fmaxf(mx, sc[i * 51 + j]);
        float sum = 0.f;
        for (int i = 0; i < 51; i++) sum += __expf(sc[i * 51 + j] - mx);
        float inv = 1.f / sum;
        for (int i = 0; i < 51; i++)
            graph[((size_t)b * 51 + i) * 51 + j] = __expf(sc[i * 51 + j] - mx) * inv;
    }
}

// ---------------------------------------------------------------------------
// K2: LSTM over L=60 steps, one wave per sequence row (B*K = 6528 rows),
// lane j owns h[j], c[j]; h broadcast via LDS each step.
// ---------------------------------------------------------------------------
__global__ __launch_bounds__(256) void lstm_kernel(
    const float* __restrict__ x,    const float* __restrict__ Wih,
    const float* __restrict__ Whh,  const float* __restrict__ bih,
    const float* __restrict__ bhh,  float* __restrict__ hbuf)
{
    __shared__ float WhhT[32 * 128];  // [k][idx] transposed for conflict-free reads
    __shared__ float WihL[128], biasL[128];
    __shared__ float hrow[8 * 32];
    int tid = threadIdx.x;

    for (int e = tid; e < 4096; e += 256) {
        int idx = e >> 5, k = e & 31;
        WhhT[k * 128 + idx] = Whh[e];
    }
    if (tid < 128) { WihL[tid] = Wih[tid]; biasL[tid] = bih[tid] + bhh[tid]; }
    __syncthreads();

    int w = tid >> 5, lane = tid & 31;
    size_t rr = (size_t)blockIdx.x * 8 + w;  // row in [0, 6528)
    float hv = 0.f, cv = 0.f;

    for (int l = 0; l < 60; l++) {
        hrow[w * 32 + lane] = hv;
        __syncthreads();
        float xt = x[rr * 60 + l];
        float z[4];
#pragma unroll
        for (int gI = 0; gI < 4; gI++) {
            int idx = gI * 32 + lane;
            float zz = biasL[idx] + WihL[idx] * xt;
#pragma unroll
            for (int k = 0; k < 32; k++) zz += WhhT[k * 128 + idx] * hrow[w * 32 + k];
            z[gI] = zz;
        }
        float ig = sigf(z[0]), fg = sigf(z[1]);
        float gg = tanhf(z[2]), og = sigf(z[3]);
        cv = fg * cv + ig * gg;
        hv = og * tanhf(cv);
        hbuf[(rr * 60 + l) * 32 + lane] = hv;
        __syncthreads();
    }
}

// ---------------------------------------------------------------------------
// K3: fused GNN: agg einsum + (agg@Wn + bn + hprev@Wr) -> relu -> @W2 + b2
// one block per (b,j); 8 waves = 4 l-tiles x 2 n-tiles; 3 WMMAs per wave.
// ---------------------------------------------------------------------------
__global__ __launch_bounds__(256) void gnn_kernel(
    const float* __restrict__ hbuf, const float* __restrict__ graph,
    const float* __restrict__ Wn,   const float* __restrict__ bn,
    const float* __restrict__ Wr,   const float* __restrict__ W2,
    const float* __restrict__ b2,   float* __restrict__ hgcnp)
{
    __shared__ float gcol[51];
    __shared__ __align__(16) unsigned short aggbf[64 * 32];
    __shared__ __align__(16) unsigned short hprevbf[64 * 32];
    __shared__ __align__(16) unsigned short hnbf[64 * 32];
    __shared__ __align__(16) unsigned short WnT[32 * 32];
    __shared__ __align__(16) unsigned short WrT[32 * 32];
    __shared__ __align__(16) unsigned short W2T[32 * 32];

    int tid = threadIdx.x;
    int b = blockIdx.x / 51, j = blockIdx.x % 51;

    if (tid < 51) gcol[tid] = graph[((size_t)b * 51 + tid) * 51 + j];
    for (int e = tid; e < 1024; e += 256) {  // weights: store column-major bf16
        int k = e >> 5, n = e & 31;
        WnT[n * 32 + k] = f2bf(Wn[e]);
        WrT[n * 32 + k] = f2bf(Wr[e]);
        W2T[n * 32 + k] = f2bf(W2[e]);
    }
    __syncthreads();

    // agg[b,j,l,:] = sum_k graph[b,k,j] * h[b,k,l,:]
    float accv[8];
#pragma unroll
    for (int p = 0; p < 8; p++) accv[p] = 0.f;
    for (int k = 0; k < 51; k++) {
        float gv = gcol[k];
        const float* hp = hbuf + ((size_t)b * 51 + k) * 60 * 32;
#pragma unroll
        for (int p = 0; p < 8; p++) {
            int e = tid + p * 256;
            int l = e >> 5, hh = e & 31;
            if (l < 60) accv[p] += hp[l * 32 + hh] * gv;
        }
    }
#pragma unroll
    for (int p = 0; p < 8; p++) {
        int e = tid + p * 256;
        int l = e >> 5, hh = e & 31;
        aggbf[e] = f2bf(l < 60 ? accv[p] : 0.f);
        float pv = (l >= 1 && l < 60)
                 ? hbuf[(((size_t)b * 51 + j) * 60 + (l - 1)) * 32 + hh] : 0.f;
        hprevbf[e] = f2bf(pv);
    }
    __syncthreads();

    int lane = tid & 31, w = tid >> 5;
    int ltile = w >> 1, ntile = w & 1;
    int half = lane >> 4, ncol = lane & 15;
    int nG = ntile * 16 + ncol;

    AFrag a, a2, bfn, bfr;
    load_afrag(a,  aggbf,   ltile * 16, lane);
    load_afrag(a2, hprevbf, ltile * 16, lane);
    load_bfrag(bfn, WnT, ntile * 16, lane);
    load_bfrag(bfr, WrT, ntile * 16, lane);

    CFrag c; czero(c);
    c.v = __builtin_amdgcn_wmma_f32_16x16x32_bf16(false, a.v,  false, bfn.v, (short)0, c.v, false, false);
    c.v = __builtin_amdgcn_wmma_f32_16x16x32_bf16(false, a2.v, false, bfr.v, (short)0, c.v, false, false);

    float bnv = bn[nG];
#pragma unroll
    for (int v = 0; v < 8; v++) {
        float val = fmaxf(c.f[v] + bnv, 0.f);
        int m = ltile * 16 + v + half * 8;
        hnbf[m * 32 + nG] = f2bf(val);
    }
    __syncthreads();

    AFrag a3, bf2;
    load_afrag(a3, hnbf, ltile * 16, lane);
    load_bfrag(bf2, W2T, ntile * 16, lane);
    CFrag c2; czero(c2);
    c2.v = __builtin_amdgcn_wmma_f32_16x16x32_bf16(false, a3.v, false, bf2.v, (short)0, c2.v, false, false);

    float b2v = b2[nG];
#pragma unroll
    for (int v = 0; v < 8; v++) {
        int m = v + half * 8;
        int l = ltile * 16 + m;
        if (l < 60) {
            size_t row = ((size_t)b * 51 + j) * 60 + l;
            hgcnp[row * 32 + nG] = c2.f[v] + b2v;
        }
    }
}

// ---------------------------------------------------------------------------
// Generic WMMA stripe GEMM: 32-column stripe, full M=128 per block
// (8 waves = 8 M-tiles). B tile (32K x 32N fp32) DMA'd into LDS by the
// Tensor Data Mover (TENSORcnt), converted bf16 in-register; A tile staged
// cooperatively as bf16. 2 WMMAs per wave per k-step (A-frag reuse).
// mode 0: relu + batchnorm store (padded ldo, zero pad cols)
// mode 2: + bias + addsrc, plain store
// mode 3: split-K partial -> atomicAdd into accumulator
// ---------------------------------------------------------------------------
__global__ __launch_bounds__(256) void gemm_stripe_kernel(
    const float* __restrict__ A, int lda, int Kreal,
    const float* __restrict__ Bm, int ldb, int Nreal,
    const float* __restrict__ bias,
    const float* __restrict__ g,  const float* __restrict__ bb,
    const float* __restrict__ mm, const float* __restrict__ vv,
    const float* __restrict__ addsrc, int ldadd,
    float* __restrict__ out, int ldo,
    int mode, int ksteps_total, int ksteps_per_block)
{
    __shared__ __align__(16) unsigned short ldsA[128 * 32];  // bf16 A tile
    __shared__ __align__(16) float ldsBf[32 * 32];           // fp32 B tile (TDM dest)

    int tid  = threadIdx.x;
    int lane = tid & 31, w = tid >> 5;
    int c0   = blockIdx.x * 32;
    int kbeg = blockIdx.y * ksteps_per_block;
    int kend = min(kbeg + ksteps_per_block, ksteps_total);

    CFrag acc0, acc1; czero(acc0); czero(acc1);

    for (int kk = kbeg; kk < kend; ++kk) {
        int k0 = kk * 32;

        // issue B-tile DMA first (overlaps with A staging below); wave 0 only
        if (w == 0) {
            int ncols = min(32, Nreal - c0);
            int nrows = min(32, Kreal - k0);
            tdm_load_tile_f32(Bm + (size_t)k0 * ldb + c0,
                              (unsigned)(uintptr_t)&ldsBf[0], ncols, nrows, ldb);
        }

        // stage A tile (128 x 32) fp32 -> bf16, packed u32 stores
        {
            int r = tid >> 1;
            int khalf = (tid & 1) << 4;
#pragma unroll
            for (int c2 = 0; c2 < 8; c2++) {
                int k = k0 + khalf + c2 * 2;
                float f0 = (k     < Kreal) ? A[(size_t)r * lda + k]     : 0.f;
                float f1 = (k + 1 < Kreal) ? A[(size_t)r * lda + k + 1] : 0.f;
                unsigned pk = (unsigned)f2bf(f0) | ((unsigned)f2bf(f1) << 16);
                *(unsigned*)&ldsA[r * 32 + khalf + c2 * 2] = pk;
            }
        }

        if (w == 0) __builtin_amdgcn_s_wait_tensorcnt(0);
        __syncthreads();

        AFrag a;
        load_afrag(a, ldsA, w * 16, lane);

        // build two B fragments from the fp32 TDM tile (row-major [k][32])
        AFrag b0f, b1f;
        int ncl = lane & 15;
        int kb  = (lane >> 4) * 16;
#pragma unroll
        for (int t = 0; t < 16; t++) {
            b0f.s[t] = f2bf(ldsBf[(kb + t) * 32 + ncl]);
            b1f.s[t] = f2bf(ldsBf[(kb + t) * 32 + 16 + ncl]);
        }

        acc0.v = __builtin_amdgcn_wmma_f32_16x16x32_bf16(
            false, a.v, false, b0f.v, (short)0, acc0.v, false, false);
        acc1.v = __builtin_amdgcn_wmma_f32_16x16x32_bf16(
            false, a.v, false, b1f.v, (short)0, acc1.v, false, false);
        __syncthreads();
    }

    // epilogue: D layout -> lane<16: N=lane, M=v; lane>=16: N=lane-16, M=8+v
    int half = lane >> 4, ncol = lane & 15;

#pragma unroll
    for (int s = 0; s < 2; s++) {
        CFrag* acc = s ? &acc1 : &acc0;
        int n = c0 + s * 16 + ncol;
        if (mode == 3) {
#pragma unroll
            for (int v = 0; v < 8; v++) {
                int m = w * 16 + v + half * 8;
                atomicAdd(&out[(size_t)m * ldo + n], acc->f[v]);
            }
        } else if (mode == 2) {
#pragma unroll
            for (int v = 0; v < 8; v++) {
                int m = w * 16 + v + half * 8;
                float val = acc->f[v] + bias[n] + addsrc[(size_t)m * ldadd + n];
                out[(size_t)m * ldo + n] = val;
            }
        } else {  // mode 0: relu + bn, padded store
#pragma unroll
            for (int v = 0; v < 8; v++) {
                int m = w * 16 + v + half * 8;
                float val = 0.f;
                if (n < Nreal) {
                    val = fmaxf(acc->f[v] + bias[n], 0.f);
                    val = (val - mm[n]) * rsqrtf(vv[n] + 1e-5f) * g[n] + bb[n];
                }
                out[(size_t)m * ldo + n] = val;
            }
        }
    }
}

// ---------------------------------------------------------------------------
// K6: ann finalize: hid = bn(relu(acc + ann_b))
// ---------------------------------------------------------------------------
__global__ void ann_fin_kernel(
    const float* __restrict__ acc, const float* __restrict__ annb,
    const float* __restrict__ g,   const float* __restrict__ bb,
    const float* __restrict__ mm,  const float* __restrict__ vv,
    float* __restrict__ hid)
{
    int i = blockIdx.x * 256 + threadIdx.x;
    if (i < 12800) {
        int m = i / 100, n = i % 100;
        float v = acc[m * 128 + n] + annb[n];
        v = fmaxf(v, 0.f);
        v = (v - mm[n]) * rsqrtf(vv[n] + 1e-5f) * g[n] + bb[n];
        hid[i] = v;
    }
}

// ---------------------------------------------------------------------------
// K7: log-prob. mask_out==0 makes the MAF affine in x with scalar constants.
// one block per batch, mean over 3060 rows.
// ---------------------------------------------------------------------------
__global__ __launch_bounds__(256) void logprob_kernel(
    const float* __restrict__ x,  const float* __restrict__ ob,
    const float* __restrict__ lg, const float* __restrict__ beta,
    float* __restrict__ lp)
{
    __shared__ float red[256];
    int tid = threadIdx.x, b = blockIdx.x;

    float m0 = ob[0], la0 = ob[1], m1 = ob[2], la1 = ob[3];
    float g0 = __expf(lg[0]), g1 = __expf(lg[1]);
    float e0 = __expf(-la0), e1 = __expf(-la1);
    float b0 = beta[0], b1 = beta[1];
    const float rs = rsqrtf(1.f + 1e-5f);
    const float cl = 0.5f * logf(1.f + 1e-5f);
    float sldj = -la0 - la1 + lg[0] + lg[1] - 2.f * cl;

    float s = 0.f;
    for (int r = tid; r < KL; r += 256) {
        float u = x[(size_t)b * KL + r];
        u = (u - m0) * e0; u = g0 * u * rs + b0;
        u = (u - m1) * e1; u = g1 * u * rs + b1;
        s += -0.5f * LOG2PI_F - 0.5f * u * u + sldj;
    }
    red[tid] = s;
    __syncthreads();
    for (int st = 128; st > 0; st >>= 1) {
        if (tid < st) red[tid] += red[tid + st];
        __syncthreads();
    }
    if (tid == 0) lp[b] = red[0] / (float)KL;
}

__global__ void mean128_kernel(const float* __restrict__ lp, float* __restrict__ outm)
{
    __shared__ float red[128];
    int tid = threadIdx.x;
    red[tid] = lp[tid];
    __syncthreads();
    for (int st = 64; st > 0; st >>= 1) {
        if (tid < st) red[tid] += red[tid + st];
        __syncthreads();
    }
    if (tid == 0) *outm = red[0] / 128.f;
}

__global__ void zero_kernel(float* __restrict__ p, int n)
{
    int i = blockIdx.x * 256 + threadIdx.x;
    if (i < n) p[i] = 0.f;
}

// ---------------------------------------------------------------------------
// launch
// ---------------------------------------------------------------------------
extern "C" void kernel_launch(void* const* d_in, const int* in_sizes, int n_in,
                              void* d_out, int out_size, void* d_ws, size_t ws_size,
                              hipStream_t stream)
{
    (void)in_sizes; (void)n_in; (void)out_size; (void)ws_size;

    const float* x    = (const float*)d_in[0];
    const float* aWq  = (const float*)d_in[1];
    const float* abq  = (const float*)d_in[2];
    const float* aWk  = (const float*)d_in[3];
    const float* abk  = (const float*)d_in[4];
    const float* lWih = (const float*)d_in[5];
    const float* lWhh = (const float*)d_in[6];
    const float* lbih = (const float*)d_in[7];
    const float* lbhh = (const float*)d_in[8];
    const float* gWn  = (const float*)d_in[9];
    const float* gbn  = (const float*)d_in[10];
    const float* gWr  = (const float*)d_in[11];
    const float* gW2  = (const float*)d_in[12];
    const float* gb2  = (const float*)d_in[13];
    const float* mW1  = (const float*)d_in[14];
    const float* mb1  = (const float*)d_in[15];
    const float* bn1g = (const float*)d_in[16];
    const float* bn1b = (const float*)d_in[17];
    const float* bn1m = (const float*)d_in[18];
    const float* bn1v = (const float*)d_in[19];
    const float* mW2  = (const float*)d_in[20];
    const float* mb2  = (const float*)d_in[21];
    const float* bn2g = (const float*)d_in[22];
    const float* bn2b = (const float*)d_in[23];
    const float* bn2m = (const float*)d_in[24];
    const float* bn2v = (const float*)d_in[25];
    const float* mW3  = (const float*)d_in[26];
    const float* mb3  = (const float*)d_in[27];
    const float* annW = (const float*)d_in[28];
    const float* annb = (const float*)d_in[29];
    const float* anng = (const float*)d_in[30];
    const float* annbb= (const float*)d_in[31];
    const float* annm = (const float*)d_in[32];
    const float* annv = (const float*)d_in[33];
    // 34..37 (maf_in_W, maf_in_b, maf_cond_W, maf_out_W) are dead (mask_out==0)
    const float* mafob   = (const float*)d_in[38];
    const float* maflg   = (const float*)d_in[39];
    const float* mafbeta = (const float*)d_in[40];

    float* ws     = (float*)d_ws;
    float* graph  = ws + WS_GRAPH;
    float* hbuf   = ws + WS_H;
    float* hgcnp  = ws + WS_HGCNP;
    float* z1     = ws + WS_Z1;
    float* z2     = ws + WS_Z2;
    float* annacc = ws + WS_ANN;

    float* out = (float*)d_out;

    // K1: attention/softmax
    attn_kernel<<<BB, 256, 0, stream>>>(x, aWq, abq, aWk, abk, graph);

    // K2: LSTM (6528 rows, 8 rows per block)
    lstm_kernel<<<816, 256, 0, stream>>>(x, lWih, lWhh, lbih, lbhh, hbuf);

    // K3: fused GNN (agg einsum + three WMMAs per wave)
    gnn_kernel<<<BB * KK_, 256, 0, stream>>>(hbuf, graph, gWn, gbn, gWr, gW2, gb2, hgcnp);

    // K4a: z1 = bn1(relu(xflat @ W1 + b1))   M=128 K=3060 N=500
    gemm_stripe_kernel<<<dim3(16, 1), 256, 0, stream>>>(
        x, KL, KL, mW1, N1, N1, mb1, bn1g, bn1b, bn1m, bn1v,
        nullptr, 0, z1, NP, 0, 96, 96);

    // K4b: z2 = bn2(relu(z1 @ W2 + b2))      M=128 K=500 N=500
    gemm_stripe_kernel<<<dim3(16, 1), 256, 0, stream>>>(
        z1, NP, N1, mW2, N1, N1, mb2, bn2g, bn2b, bn2m, bn2v,
        nullptr, 0, z2, NP, 0, 16, 16);

    // K4c: h_gcn = z2 @ W3 + b3 + h_gcn_pre  M=128 K=500 N=97920 (streams 196MB)
    gemm_stripe_kernel<<<dim3(3060, 1), 256, 0, stream>>>(
        z2, NP, N1, mW3, KL32, KL32, mb3, nullptr, nullptr, nullptr, nullptr,
        hgcnp, KL32, out + HGCN_OFF, KL32, 2, 16, 16);

    // K5: ann partials with split-K (32 splits) into accumulator
    zero_kernel<<<(128 * 128 + 255) / 256, 256, 0, stream>>>(annacc, 128 * 128);
    gemm_stripe_kernel<<<dim3(4, 32), 256, 0, stream>>>(
        out + HGCN_OFF, KL32, KL32, annW, 100, 100, nullptr,
        nullptr, nullptr, nullptr, nullptr, nullptr, 0,
        annacc, 128, 3, 3060, 96);

    // K6: hid = bn(relu(acc + ann_b))
    ann_fin_kernel<<<50, 256, 0, stream>>>(annacc, annb, anng, annbb, annm, annv,
                                           out + HID_OFF);

    // K7/K8: log-prob (MAF collapsed to affine transform) and its mean
    logprob_kernel<<<BB, 256, 0, stream>>>(x, mafob, maflg, mafbeta, out + LP_OFF);
    mean128_kernel<<<1, 128, 0, stream>>>(out + LP_OFF, out + LPM_OFF);
}
